// SAERTextRNN_91225105367293
// MI455X (gfx1250) — matrix operand order repaired and checked by reading, not compiled, not run on hardware
//
#include <hip/hip_runtime.h>

#define SDIM 32
#define BDIM 32
#define VDIM 32000
#define HDIM 512
#define DINDIM 512
#define FDIM 50

typedef __attribute__((ext_vector_type(16))) __bf16 v16bf;
typedef __attribute__((ext_vector_type(8)))  __bf16 v8bf;
typedef __attribute__((ext_vector_type(8)))  float  v8f;

// float -> bf16, round-to-nearest-even
__device__ __forceinline__ __bf16 f2bf(float f) {
  union { float f; unsigned u; } v; v.f = f;
  unsigned r = (v.u + 0x7FFFu + ((v.u >> 16) & 1u)) >> 16;
  unsigned short s = (unsigned short)r;
  __bf16 o; __builtin_memcpy(&o, &s, 2);
  return o;
}

// A fragment (16-bit A 16x32 layout, ISA 7.12.2): per lane, elements 0..7 hold
// K = 8*half+0..7 and elements 8..15 hold K = 16+8*half+0..7 -> two b128 loads.
__device__ __forceinline__ v16bf ld_afrag(const __bf16* __restrict__ p, int half) {
  v8bf lo = *(const v8bf*)(p + 8 * half);
  v8bf hi = *(const v8bf*)(p + 16 + 8 * half);
  return __builtin_shufflevector(lo, hi, 0, 1, 2, 3, 4, 5, 6, 7, 8, 9, 10, 11, 12, 13, 14, 15);
}

// B fragment (16-bit B 32x16): per lane, elements 0..7 hold K = 8*half+0..7 and
// elements 8..15 hold K = 16+8*half+0..7 -> two b128 loads.
__device__ __forceinline__ v16bf ld_bfrag(const __bf16* __restrict__ p, int half) {
  v8bf lo = *(const v8bf*)(p + 8 * half);
  v8bf hi = *(const v8bf*)(p + 16 + 8 * half);
  return __builtin_shufflevector(lo, hi, 0, 1, 2, 3, 4, 5, 6, 7, 8, 9, 10, 11, 12, 13, 14, 15);
}

// C(MxN) = A(MxK bf16, row-major) * W(NxK bf16, row-major)^T + bias(N)
// One wave per 32x(16*NT) C block: 2 M-tiles x NT N-tiles, 2*NT WMMAs per k-step.
// Requires M % 32 == 0, N % (16*NT) == 0, K % 32 == 0 (all shapes here comply).
template <int NT>
__global__ __launch_bounds__(32)
void k_gemm(const __bf16* __restrict__ A, const __bf16* __restrict__ W,
            const float* __restrict__ bias, float* __restrict__ C, int N, int K) {
  const int lane = threadIdx.x;      // 0..31
  const int half = lane >> 4;        // 0 or 1
  const int lq   = lane & 15;        // 0..15
  const int n0 = blockIdx.x * (16 * NT);
  const int m0 = blockIdx.y * 32;

  v8f acc[2][NT] = {};
  const __bf16* Ar0 = A + (size_t)(m0 + lq) * K;
  const __bf16* Ar1 = A + (size_t)(m0 + 16 + lq) * K;

  for (int kb = 0; kb < K; kb += 32) {
    v16bf a0 = ld_afrag(Ar0 + kb, half);
    v16bf a1 = ld_afrag(Ar1 + kb, half);
#pragma unroll
    for (int nt = 0; nt < NT; ++nt) {
      const __bf16* Wr = W + (size_t)(n0 + nt * 16 + lq) * K + kb;
      v16bf b = ld_bfrag(Wr, half);
      acc[0][nt] = __builtin_amdgcn_wmma_f32_16x16x32_bf16(
          false, a0, false, b, (short)0, acc[0][nt], false, false);
      acc[1][nt] = __builtin_amdgcn_wmma_f32_16x16x32_bf16(
          false, a1, false, b, (short)0, acc[1][nt], false, false);
    }
  }

  // C layout: VGPR i, lanes 0-15 -> M=i, lanes 16-31 -> M=i+8; N = lane&15
#pragma unroll
  for (int mt = 0; mt < 2; ++mt)
#pragma unroll
    for (int nt = 0; nt < NT; ++nt)
#pragma unroll
      for (int i = 0; i < 8; ++i) {
        int m = m0 + mt * 16 + i + 8 * half;
        int n = n0 + nt * 16 + lq;
        C[(size_t)m * N + n] = acc[mt][nt][i] + bias[n];
      }
}

// f32 -> bf16 bulk convert
__global__ void k_cvt(const float* __restrict__ src, __bf16* __restrict__ dst, int n) {
  int i = blockIdx.x * blockDim.x + threadIdx.x;
  if (i < n) dst[i] = f2bf(src[i]);
}

// out_bf[r, d] = bf16(table[ids[r], d])
__global__ void k_gather_bf16(const float* __restrict__ table, const int* __restrict__ ids,
                              __bf16* __restrict__ out, int rows, int dim) {
  int i = blockIdx.x * blockDim.x + threadIdx.x;
  if (i >= rows * dim) return;
  int r = i / dim, d = i - r * dim;
  out[i] = f2bf(table[(size_t)ids[r] * dim + d]);
}

// dstf = src (f32), dstb = bf16(src)
__global__ void k_copy_cvt(const float* __restrict__ src, float* __restrict__ dstf,
                           __bf16* __restrict__ dstb, int n) {
  int i = blockIdx.x * blockDim.x + threadIdx.x;
  if (i < n) { float x = src[i]; dstf[i] = x; dstb[i] = f2bf(x); }
}

// GRU gate math for one time step.
// gi: (S*B, 3H) precomputed; gh: (B, 3H); hf/hb: (B, H) state f32/bf16 in-out;
// seqf/seqb: (S*B, H) per-step outputs.
__global__ void k_gru_gates(const float* __restrict__ gi, const float* __restrict__ gh,
                            float* __restrict__ hf, __bf16* __restrict__ hb,
                            float* __restrict__ seqf, __bf16* __restrict__ seqb, int t) {
  int i = blockIdx.x * blockDim.x + threadIdx.x;
  if (i >= BDIM * HDIM) return;
  int b = i / HDIM, c = i - b * HDIM;
  const float* gir = gi + (size_t)(t * BDIM + b) * (3 * HDIM);
  const float* ghr = gh + (size_t)b * (3 * HDIM);
  float ir = gir[c], iz = gir[HDIM + c], inn = gir[2 * HDIM + c];
  float hr = ghr[c], hz = ghr[HDIM + c], hn  = ghr[2 * HDIM + c];
  float r = 1.0f / (1.0f + __expf(-(ir + hr)));
  float z = 1.0f / (1.0f + __expf(-(iz + hz)));
  float n = tanhf(inn + r * hn);
  float h2 = (1.0f - z) * n + z * hf[i];
  hf[i] = h2;
  hb[i] = f2bf(h2);
  size_t so = (size_t)(t * BDIM + b) * HDIM + c;
  seqf[so] = h2;
  seqb[so] = f2bf(h2);
}

// fg/rg: per (s,b) row, two H-length dots + sigmoid
__global__ void k_fg_rg(const float* __restrict__ outseq,
                        const float* __restrict__ Wfg, const float* __restrict__ bfg,
                        const float* __restrict__ Wrg, const float* __restrict__ brg,
                        float* __restrict__ fg, float* __restrict__ rg) {
  int m = blockIdx.x * blockDim.x + threadIdx.x;
  if (m >= SDIM * BDIM) return;
  const float* r = outseq + (size_t)m * HDIM;
  float a = 0.f, bacc = 0.f;
  for (int i = 0; i < HDIM; ++i) { float x = r[i]; a += x * Wfg[i]; bacc += x * Wrg[i]; }
  fg[m] = 1.0f / (1.0f + __expf(-(a + bfg[0])));
  rg[m] = 1.0f / (1.0f + __expf(-(bacc + brg[0])));
}

// out2 = bf16(tanh(tmp + rg[m] * score_out[b, :]))
__global__ void k_oomix(const float* __restrict__ tmp, const float* __restrict__ rg,
                        const float* __restrict__ so, __bf16* __restrict__ out2) {
  int i = blockIdx.x * blockDim.x + threadIdx.x;
  if (i >= SDIM * BDIM * HDIM) return;
  int m = i / HDIM, c = i - m * HDIM;
  int b = m % BDIM;
  out2[i] = f2bf(tanhf(tmp[i] + rg[m] * so[b * HDIM + c]));
}

// Per (s,b): scores over F features (dot-80 vs energy), mask, softmax -> f_attn
__global__ __launch_bounds__(64)
void k_attn(const float* __restrict__ fs, const float* __restrict__ user,
            const float* __restrict__ item, const float* __restrict__ energy,
            const unsigned char* __restrict__ mask, float* __restrict__ fattn) {
  int row = blockIdx.x;            // s*B + b
  int b = row % BDIM;
  int t = threadIdx.x;             // 0..63
  __shared__ float q[80];
  __shared__ float sc[64];
  __shared__ float red[64];
  for (int j = t; j < 80; j += 64)
    q[j] = (j < 16) ? fs[(size_t)row * 16 + j]
                    : (j < 48 ? user[b * 32 + (j - 16)] : item[b * 32 + (j - 48)]);
  __syncthreads();
  float s = -INFINITY;
  if (t < FDIM) {
    const float* e = energy + ((size_t)b * FDIM + t) * 80;
    float acc = 0.f;
    for (int j = 0; j < 80; ++j) acc += q[j] * e[j];
    s = mask[b * FDIM + t] ? acc : -1e9f;
  }
  sc[t] = s;
  red[t] = s;
  __syncthreads();
  for (int k = 32; k > 0; k >>= 1) {
    if (t < k) red[t] = fmaxf(red[t], red[t + k]);
    __syncthreads();
  }
  float mx = red[0];
  __syncthreads();
  float ex = (t < FDIM) ? __expf(sc[t] - mx) : 0.f;
  red[t] = ex;
  __syncthreads();
  for (int k = 32; k > 0; k >>= 1) {
    if (t < k) red[t] += red[t + k];
    __syncthreads();
  }
  if (t < FDIM) fattn[(size_t)row * FDIM + t] = ex / red[0];
}

// In-place over d_out row (V=32000): softmax, mix with scattered feature attention, log.
__global__ __launch_bounds__(256)
void k_final(float* __restrict__ out, const float* __restrict__ fg,
             const float* __restrict__ fattn, const int* __restrict__ ifeat) {
  int row = blockIdx.x;            // s*B + b
  int b = row % BDIM;
  float* p = out + (size_t)row * VDIM;
  __shared__ float red[256];
  int t = threadIdx.x;

  float mx = -INFINITY;
  for (int v = t; v < VDIM; v += 256) mx = fmaxf(mx, p[v]);
  red[t] = mx; __syncthreads();
  for (int k = 128; k > 0; k >>= 1) { if (t < k) red[t] = fmaxf(red[t], red[t + k]); __syncthreads(); }
  mx = red[0]; __syncthreads();

  float sum = 0.f;
  for (int v = t; v < VDIM; v += 256) sum += __expf(p[v] - mx);
  red[t] = sum; __syncthreads();
  for (int k = 128; k > 0; k >>= 1) { if (t < k) red[t] += red[t + k]; __syncthreads(); }
  sum = red[0]; __syncthreads();

  float g = fg[row];
  float scale = (1.0f - g) / sum;
  for (int v = t; v < VDIM; v += 256) p[v] = scale * __expf(p[v] - mx);
  __syncthreads();
  if (t < FDIM) {
    int idx = ifeat[b * FDIM + t];                  // unique per row
    p[idx] += g * fattn[(size_t)row * FDIM + t];
  }
  __syncthreads();
  for (int v = t; v < VDIM; v += 256) p[v] = __logf(p[v]);
}

extern "C" void kernel_launch(void* const* d_in, const int* in_sizes, int n_in,
                              void* d_out, int out_size, void* d_ws, size_t ws_size,
                              hipStream_t stream) {
  (void)in_sizes; (void)n_in; (void)out_size; (void)ws_size;
  const int*   input_seq   = (const int*)d_in[0];
  const float* init_hidden = (const float*)d_in[1];
  const float* user_vct    = (const float*)d_in[2];
  const float* item_vct    = (const float*)d_in[3];
  const float* ui_var      = (const float*)d_in[4];
  const int*   i_features  = (const int*)d_in[5];
  const unsigned char* if_mask = (const unsigned char*)d_in[6];
  const float* word_ebd    = (const float*)d_in[7];
  const float* Wih0 = (const float*)d_in[8],  *Whh0 = (const float*)d_in[9];
  const float* bih0 = (const float*)d_in[10], *bhh0 = (const float*)d_in[11];
  const float* Wih1 = (const float*)d_in[12], *Whh1 = (const float*)d_in[13];
  const float* bih1 = (const float*)d_in[14], *bhh1 = (const float*)d_in[15];
  const float* Wout = (const float*)d_in[16], *bout = (const float*)d_in[17];
  const float* Wrg  = (const float*)d_in[18], *brg  = (const float*)d_in[19];
  const float* Wso  = (const float*)d_in[20], *bso  = (const float*)d_in[21];
  const float* Woo  = (const float*)d_in[22], *boo  = (const float*)d_in[23];
  const float* Wfg  = (const float*)d_in[24], *bfg  = (const float*)d_in[25];
  const float* Wfs  = (const float*)d_in[26], *bfs  = (const float*)d_in[27];
  const float* Wfa  = (const float*)d_in[28], *bfa  = (const float*)d_in[29];

  const int SB = SDIM * BDIM;           // 1024
  const int H3 = 3 * HDIM;              // 1536

  // ---- workspace layout: f32 region then bf16 region (16B aligned) ----
  float* wf = (float*)d_ws;
  size_t o = 0;
  float* gi0      = wf + o; o += (size_t)SB * H3;          // 1572864
  float* gi1      = wf + o; o += (size_t)SB * H3;          // 1572864
  float* gh       = wf + o; o += (size_t)BDIM * H3;        // 49152
  float* hbuf     = wf + o; o += (size_t)BDIM * HDIM;      // 16384
  float* h0seqf   = wf + o; o += (size_t)SB * HDIM;        // 524288
  float* outseqf  = wf + o; o += (size_t)SB * HDIM;        // 524288
  float* fg       = wf + o; o += (size_t)SB;               // 1024
  float* rg       = wf + o; o += (size_t)SB;               // 1024
  float* fs       = wf + o; o += (size_t)SB * 16;          // 16384
  float* scoreout = wf + o; o += (size_t)BDIM * HDIM;      // 16384
  float* tmpoo    = wf + o; o += (size_t)SB * HDIM;        // 524288
  float* energy   = wf + o; o += (size_t)BDIM * FDIM * 80; // 128000
  float* fattn    = wf + o; o += (size_t)SB * FDIM;        // 51200

  __bf16* wb = (__bf16*)(wf + o);
  size_t b = 0;
  __bf16* emb_b    = wb + b; b += (size_t)SB * DINDIM;          // activations
  __bf16* hbuf_b   = wb + b; b += (size_t)BDIM * HDIM;
  __bf16* h0seq_b  = wb + b; b += (size_t)SB * HDIM;
  __bf16* outseq_b = wb + b; b += (size_t)SB * HDIM;
  __bf16* out2_b   = wb + b; b += (size_t)SB * HDIM;
  __bf16* uivar_b  = wb + b; b += (size_t)BDIM * 32;
  __bf16* feats_b  = wb + b; b += (size_t)BDIM * FDIM * DINDIM;
  __bf16* Wih0_b   = wb + b; b += (size_t)H3 * DINDIM;          // weights
  __bf16* Whh0_b   = wb + b; b += (size_t)H3 * HDIM;
  __bf16* Wih1_b   = wb + b; b += (size_t)H3 * HDIM;
  __bf16* Whh1_b   = wb + b; b += (size_t)H3 * HDIM;
  __bf16* Woo_b    = wb + b; b += (size_t)HDIM * HDIM;
  __bf16* Wso_b    = wb + b; b += (size_t)HDIM * 32;
  __bf16* Wfs_b    = wb + b; b += (size_t)16 * HDIM;
  __bf16* Wfa_b    = wb + b; b += (size_t)80 * DINDIM;
  __bf16* Wout_b   = wb + b; b += (size_t)VDIM * HDIM;
  // total workspace ~66 MB

  float* logits = (float*)d_out;        // (S*B, V), transformed in place

#define CVT(src, dst, n) k_cvt<<<((n) + 255) / 256, 256, 0, stream>>>(src, dst, n)
  // ---- 0) one-shot weight/operand conversions to bf16 ----
  CVT(Wih0, Wih0_b, H3 * DINDIM);
  CVT(Whh0, Whh0_b, H3 * HDIM);
  CVT(Wih1, Wih1_b, H3 * HDIM);
  CVT(Whh1, Whh1_b, H3 * HDIM);
  CVT(Woo,  Woo_b,  HDIM * HDIM);
  CVT(Wso,  Wso_b,  HDIM * 32);
  CVT(Wfs,  Wfs_b,  16 * HDIM);
  CVT(Wfa,  Wfa_b,  80 * DINDIM);
  CVT(Wout, Wout_b, VDIM * HDIM);
  CVT(ui_var, uivar_b, BDIM * 32);
#undef CVT

  // ---- 1) embedding gather straight to bf16 ----
  k_gather_bf16<<<(SB * DINDIM + 255) / 256, 256, 0, stream>>>(
      word_ebd, input_seq, emb_b, SB, DINDIM);

  // ---- 2) gi0 = emb @ Wih0^T + bih0  (1024 x 1536, K=512) ----
  k_gemm<4><<<dim3(H3 / 64, SB / 32), 32, 0, stream>>>(emb_b, Wih0_b, bih0, gi0, H3, DINDIM);

  // ---- 3) GRU layer 0 ----
  k_copy_cvt<<<(BDIM * HDIM + 255) / 256, 256, 0, stream>>>(
      init_hidden, hbuf, hbuf_b, BDIM * HDIM);
  for (int t = 0; t < SDIM; ++t) {
    k_gemm<4><<<dim3(H3 / 64, 1), 32, 0, stream>>>(hbuf_b, Whh0_b, bhh0, gh, H3, HDIM);
    k_gru_gates<<<(BDIM * HDIM + 255) / 256, 256, 0, stream>>>(
        gi0, gh, hbuf, hbuf_b, h0seqf, h0seq_b, t);
  }

  // ---- 4) gi1 = h0seq @ Wih1^T + bih1 ----
  k_gemm<4><<<dim3(H3 / 64, SB / 32), 32, 0, stream>>>(h0seq_b, Wih1_b, bih1, gi1, H3, HDIM);

  // ---- 5) GRU layer 1 ----
  k_copy_cvt<<<(BDIM * HDIM + 255) / 256, 256, 0, stream>>>(
      init_hidden + BDIM * HDIM, hbuf, hbuf_b, BDIM * HDIM);
  for (int t = 0; t < SDIM; ++t) {
    k_gemm<4><<<dim3(H3 / 64, 1), 32, 0, stream>>>(hbuf_b, Whh1_b, bhh1, gh, H3, HDIM);
    k_gru_gates<<<(BDIM * HDIM + 255) / 256, 256, 0, stream>>>(
        gi1, gh, hbuf, hbuf_b, outseqf, outseq_b, t);
  }

  // ---- 6) fg / rg scalar gates ----
  k_fg_rg<<<(SB + 255) / 256, 256, 0, stream>>>(outseqf, Wfg, bfg, Wrg, brg, fg, rg);

  // ---- 7) fs = outseq @ Wfs^T + bfs  (1024 x 16) ----
  k_gemm<1><<<dim3(1, SB / 32), 32, 0, stream>>>(outseq_b, Wfs_b, bfs, fs, 16, HDIM);

  // ---- 8) score_out = ui_var @ Wso^T + bso  (32 x 512, K=32) ----
  k_gemm<4><<<dim3(HDIM / 64, 1), 32, 0, stream>>>(uivar_b, Wso_b, bso, scoreout, HDIM, 32);

  // ---- 9) tmpoo = outseq @ Woo^T + boo; out2 = bf16(tanh(tmpoo + rg*score_out)) ----
  k_gemm<4><<<dim3(HDIM / 64, SB / 32), 32, 0, stream>>>(outseq_b, Woo_b, boo, tmpoo, HDIM, HDIM);
  k_oomix<<<(SB * HDIM + 255) / 256, 256, 0, stream>>>(tmpoo, rg, scoreout, out2_b);

  // ---- 10) features gather (bf16); energy = features @ Wfa^T + bfa (1600 x 80) ----
  k_gather_bf16<<<(BDIM * FDIM * DINDIM + 255) / 256, 256, 0, stream>>>(
      word_ebd, i_features, feats_b, BDIM * FDIM, DINDIM);
  k_gemm<1><<<dim3(80 / 16, BDIM * FDIM / 32), 32, 0, stream>>>(
      feats_b, Wfa_b, bfa, energy, 80, DINDIM);

  // ---- 11) attention scores + masked softmax over F ----
  k_attn<<<SB, 64, 0, stream>>>(fs, user_vct, item_vct, energy, if_mask, fattn);

  // ---- 12) logits = out2 @ Wout^T + bout  (1024 x 32000) -> d_out ----
  k_gemm<4><<<dim3(VDIM / 64, SB / 32), 32, 0, stream>>>(out2_b, Wout_b, bout, logits, VDIM, HDIM);

  // ---- 13) in-place softmax over V, mix scattered feature attention, log ----
  k_final<<<SB, 256, 0, stream>>>(logits, fg, fattn, i_features);
}
